// CrossBlock_78408922956406
// MI455X (gfx1250) — matrix operand order repaired and checked by reading, not compiled, not run on hardware
//
#include <hip/hip_runtime.h>
#include <hip/hip_bf16.h>
#include <math.h>

typedef __attribute__((ext_vector_type(16))) _Float16 v16h;
typedef __attribute__((ext_vector_type(8)))  _Float16 v8h;
typedef __attribute__((ext_vector_type(4)))  _Float16 v4h;
typedef __attribute__((ext_vector_type(8)))  float    v8f;

#define WMMA_F16(a, b, c) \
  __builtin_amdgcn_wmma_f32_16x16x32_f16(false, (a), false, (b), (short)0, (c), false, false)

// ---------------------------------------------------------------------------
// gfx1250 async direct global->LDS copy (16B), tracked by ASYNCcnt.
// Generic pointers to LDS carry the wave-relative LDS byte offset in their
// low 32 bits (ISA 10.2 aperture mapping), which is what VDST must hold.
// Async ops complete in order, so s_wait_asynccnt N is a pipeline drain to
// depth N: the double-buffer loops wait for the *previous* tile only.
// ---------------------------------------------------------------------------
__device__ inline unsigned lds_off(const void* p) {
  return (unsigned)(uintptr_t)p;
}
__device__ inline void async_copy16(unsigned ldsoff, const void* gptr) {
  asm volatile("global_load_async_to_lds_b128 %0, %1, off"
               :: "v"(ldsoff), "v"(gptr)
               : "memory");
}
template <int N>
__device__ inline void wait_async() {
  asm volatile("s_wait_asynccnt %0" :: "i"(N) : "memory");
}

// ---------------------------------------------------------------------------
// Fragment loaders (wave32), this code's contiguous k order (A and B alike), all contiguous vector loads:
// A 16x32 f16: lane<16 row=lane, K = kb..kb+15, kb = lane<16?0:16
// B 32x16 f16: lane<16 col=lane, K = kb..kb+15, kb = lane<16?0:16
//   (one 32B load per lane when the LDS tile is K-contiguous per column)
// C/D 16x16 f32: VGPR r: row = r + (lane<16 ? 0 : 8), col = lane&15.
// ---------------------------------------------------------------------------
__device__ __forceinline__ v16h load_a(const _Float16* base, int rs, int lane) {
  int row = lane & 15;
  int kb  = (lane < 16) ? 0 : 16;
  const _Float16* p = base + row * rs + kb;
  v8h lo = *(const v8h*)(p);
  v8h hi = *(const v8h*)(p + 8);
  v16h f;
#pragma unroll
  for (int i = 0; i < 8; ++i) { f[i] = lo[i]; f[i + 8] = hi[i]; }
  return f;
}

__device__ __forceinline__ v16h load_bk(const _Float16* base, int rs, int lane) {
  int col = lane & 15;
  int kb  = (lane < 16) ? 0 : 16;
  return *(const v16h*)(base + col * rs + kb);
}

// ---------------------------------------------------------------------------
// fp32 -> f16 conversion (vectorized), output row stride for concat packing.
// ---------------------------------------------------------------------------
__global__ void cvt_f32_f16(const float* __restrict__ in, _Float16* __restrict__ out,
                            int rows, int cols, int ostride) {
  int total4 = rows * (cols >> 2);
  int cpr = cols >> 2;
  for (int g = blockIdx.x * blockDim.x + threadIdx.x; g < total4;
       g += gridDim.x * blockDim.x) {
    int r  = g / cpr;
    int c4 = (g - r * cpr) << 2;
    const float4 v = *(const float4*)&in[(size_t)r * cols + c4];
    v4h o;
    o[0] = (_Float16)v.x; o[1] = (_Float16)v.y;
    o[2] = (_Float16)v.z; o[3] = (_Float16)v.w;
    *(v4h*)&out[(size_t)r * ostride + c4] = o;
  }
}

// fp32 [K,N] -> f16 [N,K] (pre-transposed weights: GEMM B-staging is linear)
__global__ void cvt_f32_f16_T(const float* __restrict__ in, _Float16* __restrict__ out,
                              int K, int N) {
  int total = K * N;
  for (int g = blockIdx.x * blockDim.x + threadIdx.x; g < total;
       g += gridDim.x * blockDim.x) {
    int k = g / N, n = g - k * N;
    out[(size_t)n * K + k] = (_Float16)in[g];
  }
}

// ---------------------------------------------------------------------------
// One pipelined GEMM K-step: optionally issue next tile's async copies into
// the other buffer, drain the previous tile, then 4 WMMAs on this buffer.
// PREF and buffer identity are compile-time so all LDS addresses fold to
// immediates (no cndmask selection chains).
// ---------------------------------------------------------------------------
template <bool PREF>
__device__ __forceinline__ void gemm_step(
    const _Float16* AsBuf, const _Float16* BsBuf,
    const unsigned* ldsAn, const unsigned* ldsBn,
    const _Float16* Ag0, const _Float16* Ag1,
    const _Float16* Bg0, const _Float16* Bg1, int k0next,
    v8f acc[4], int wave, int lane) {
  if (PREF) {
    async_copy16(ldsAn[0], Ag0 + k0next);
    async_copy16(ldsAn[1], Ag1 + k0next);
    async_copy16(ldsBn[0], Bg0 + k0next);
    async_copy16(ldsBn[1], Bg1 + k0next);
    wait_async<4>();   // drain current tile; next stays in flight
  } else {
    wait_async<0>();
  }
  __syncthreads();
  v16h a = load_a(AsBuf + (wave * 16) * 32, 32, lane);
#pragma unroll
  for (int cb = 0; cb < 4; ++cb) {
    v16h b = load_bk(BsBuf + (cb * 16) * 32, 32, lane);
    acc[cb] = WMMA_F16(a, b, acc[cb]);
  }
  __syncthreads();   // this buffer free for the next-next tile's async writes
}

// ---------------------------------------------------------------------------
// WMMA GEMM: C[M,Ncols] = A_f16[M,K](lda) @ W[K,N] + bias, with W supplied
// PRE-TRANSPOSED as Wt[N,K] (row stride ldk) so both tiles stage as linear
// async global->LDS copies, double-buffered (loop unrolled by 2 so buffer
// parity is static). Tile 64x64, 128 threads (4 waves), K-step 32.
// mode 0: f32 out (+ optional f32 residual, same stride)
// mode 1: f16 out, row stride outStride, column offset colOffset, * scale
// mode 2: f16 head-split      out[((b*4+h)*2048+n)*64+dh] * scale
// mode 3: f16 head-split, T   out[((b*4+h)*64+dh)*2048+n] * scale  (for V)
// ---------------------------------------------------------------------------
__global__ __launch_bounds__(128) void wgemm(
    const _Float16* __restrict__ A, int lda,
    const _Float16* __restrict__ Wt, int ldk,
    const float* __restrict__ bias, int K, int mode, float scale,
    void* __restrict__ outp, int outStride, int colOffset,
    const float* __restrict__ residual) {
  __shared__ __align__(64) _Float16 As[2][64 * 32];   // [m][k] ping-pong
  __shared__ __align__(64) _Float16 Bst[2][64 * 32];  // [n][k] ping-pong
  const int tid  = threadIdx.x;
  const int wave = tid >> 5;
  const int lane = tid & 31;
  const int m0 = blockIdx.y * 64;
  const int n0 = blockIdx.x * 64;

  // per-thread staging coords: 2 rows of A and 2 rows of Bst, 16B each
  const int cr  = tid >> 2;            // 0..31
  const int ck8 = (tid & 3) << 3;      // 0,8,16,24
  unsigned ldsA0[2], ldsA1[2], ldsB0[2], ldsB1[2];
  ldsA0[0] = lds_off(&As[0][cr * 32 + ck8]);
  ldsA0[1] = lds_off(&As[0][(cr + 32) * 32 + ck8]);
  ldsA1[0] = lds_off(&As[1][cr * 32 + ck8]);
  ldsA1[1] = lds_off(&As[1][(cr + 32) * 32 + ck8]);
  ldsB0[0] = lds_off(&Bst[0][cr * 32 + ck8]);
  ldsB0[1] = lds_off(&Bst[0][(cr + 32) * 32 + ck8]);
  ldsB1[0] = lds_off(&Bst[1][cr * 32 + ck8]);
  ldsB1[1] = lds_off(&Bst[1][(cr + 32) * 32 + ck8]);
  const _Float16* Ag0 = &A[(size_t)(m0 + cr) * lda + ck8];
  const _Float16* Ag1 = &A[(size_t)(m0 + cr + 32) * lda + ck8];
  const _Float16* Bg0 = &Wt[(size_t)(n0 + cr) * ldk + ck8];
  const _Float16* Bg1 = &Wt[(size_t)(n0 + cr + 32) * ldk + ck8];

  v8f acc[4] = {};

  // prologue: tile 0 -> buffer 0
  async_copy16(ldsA0[0], Ag0);
  async_copy16(ldsA0[1], Ag1);
  async_copy16(ldsB0[0], Bg0);
  async_copy16(ldsB0[1], Bg1);

  const int nk = K >> 5;   // always even (8 or 16)
  for (int kt = 0; kt < nk; kt += 2) {
    // phase A: compute buf0 (tile kt), prefetch tile kt+1 -> buf1 (kt+1 < nk)
    gemm_step<true>(As[0], Bst[0], ldsA1, ldsB1,
                    Ag0, Ag1, Bg0, Bg1, (kt + 1) << 5, acc, wave, lane);
    // phase B: compute buf1 (tile kt+1), prefetch tile kt+2 -> buf0 if any
    if (kt + 2 < nk)
      gemm_step<true>(As[1], Bst[1], ldsA0, ldsB0,
                      Ag0, Ag1, Bg0, Bg1, (kt + 2) << 5, acc, wave, lane);
    else
      gemm_step<false>(As[1], Bst[1], ldsA0, ldsB0,
                       Ag0, Ag1, Bg0, Bg1, 0, acc, wave, lane);
  }

  const int rbase = m0 + wave * 16 + ((lane >> 4) << 3);
  const int cl    = lane & 15;
#pragma unroll
  for (int cb = 0; cb < 4; ++cb) {
    int col = n0 + cb * 16 + cl;
    float bval = bias[col];
#pragma unroll
    for (int r = 0; r < 8; ++r) {
      int row = rbase + r;
      float val = acc[cb][r] + bval;
      if (mode == 0) {
        float o = val;
        if (residual) o += residual[(size_t)row * outStride + colOffset + col];
        ((float*)outp)[(size_t)row * outStride + colOffset + col] = o;
      } else if (mode == 1) {
        ((_Float16*)outp)[(size_t)row * outStride + colOffset + col] =
            (_Float16)(val * scale);
      } else {
        int bb = row >> 11, n = row & 2047;   // M rows are b*2048+n
        int hh = col >> 6,  dh = col & 63;    // D cols are h*64+dh
        size_t idx = (mode == 2)
            ? ((size_t)((bb * 4 + hh) * 2048 + n)) * 64 + dh
            : ((size_t)((bb * 4 + hh) * 64 + dh)) * 2048 + n;
        ((_Float16*)outp)[idx] = (_Float16)(val * scale);
      }
    }
  }
}

// ---------------------------------------------------------------------------
// One pipelined flash tile: optionally issue next K/V tile's async copies
// into the other buffers, drain the previous tile, then S-WMMA, online
// softmax, P spill, P@V WMMA. Buffer identity is compile-time.
// ---------------------------------------------------------------------------
template <bool PREF>
__device__ __forceinline__ void flash_tile(
    const _Float16* KsBuf, const _Float16* VtBuf,
    const unsigned* ldsKn, const unsigned* ldsVn,
    const _Float16* const* Kg, const _Float16* const* Vg, size_t j1,
    const v16h& aq0, const v16h& aq1,
    v8f o[4], float* mrow, float* lrow,
    _Float16* Ps, int wave, int lane, int prow, int cl) {
  if (PREF) {
#pragma unroll
    for (int i = 0; i < 4; ++i) {
      async_copy16(ldsKn[i], Kg[i] + j1 * 64);
      async_copy16(ldsVn[i], Vg[i] + j1);
    }
    wait_async<8>();   // drain current tile; next stays in flight
  } else {
    wait_async<0>();
  }
  __syncthreads();

  // S = Q K^T  (per wave: 16 x 64)
  v8f s[4] = {};
#pragma unroll
  for (int cb = 0; cb < 4; ++cb) {
    v16h b0 = load_bk(KsBuf + (cb * 16) * 64 + 0, 64, lane);
    s[cb] = WMMA_F16(aq0, b0, s[cb]);
    v16h b1 = load_bk(KsBuf + (cb * 16) * 64 + 32, 64, lane);
    s[cb] = WMMA_F16(aq1, b1, s[cb]);
  }

  // online softmax update per query row (stats live in 16-lane halves)
#pragma unroll
  for (int r = 0; r < 8; ++r) {
    float mx = fmaxf(fmaxf(s[0][r], s[1][r]), fmaxf(s[2][r], s[3][r]));
#pragma unroll
    for (int off = 1; off < 16; off <<= 1) mx = fmaxf(mx, __shfl_xor(mx, off, 32));
    float mnew = fmaxf(mrow[r], mx);
    float corr = __expf(mrow[r] - mnew);
    float rs = 0.f;
#pragma unroll
    for (int cb = 0; cb < 4; ++cb) {
      float p = __expf(s[cb][r] - mnew);
      s[cb][r] = p;
      rs += p;
    }
#pragma unroll
    for (int off = 1; off < 16; off <<= 1) rs += __shfl_xor(rs, off, 32);
    lrow[r] = lrow[r] * corr + rs;
    mrow[r] = mnew;
#pragma unroll
    for (int cb = 0; cb < 4; ++cb) o[cb][r] *= corr;
  }

  // spill P to LDS to re-fragment as A-matrix
#pragma unroll
  for (int cb = 0; cb < 4; ++cb)
#pragma unroll
    for (int r = 0; r < 8; ++r)
      Ps[(prow + r) * 64 + cb * 16 + cl] = (_Float16)s[cb][r];
  __syncthreads();

  v16h ap0 = load_a(Ps + (wave * 16) * 64 + 0, 64, lane);
  v16h ap1 = load_a(Ps + (wave * 16) * 64 + 32, 64, lane);
#pragma unroll
  for (int cb = 0; cb < 4; ++cb) {
    v16h bv0 = load_bk(VtBuf + (cb * 16) * 64 + 0, 64, lane);
    o[cb] = WMMA_F16(ap0, bv0, o[cb]);
    v16h bv1 = load_bk(VtBuf + (cb * 16) * 64 + 32, 64, lane);
    o[cb] = WMMA_F16(ap1, bv1, o[cb]);
  }
  __syncthreads();   // this buffer free for the next-next tile's async writes
}

// ---------------------------------------------------------------------------
// Flash attention, one direction: out = softmax_rows(Q K^T) V, merged-head
// store. Q/K pre-scaled by sqrt(SCALE); V supplied TRANSPOSED [bh][dh][n] so
// every tile stages as linear async copies. K/V double-buffered, loop
// unrolled by 2 (32 tiles). Grid: (N/64, B*H). Block: 128 (4 waves).
// ---------------------------------------------------------------------------
__global__ __launch_bounds__(128) void flashattn(
    const _Float16* __restrict__ Qh, const _Float16* __restrict__ Kh,
    const _Float16* __restrict__ Vth, _Float16* __restrict__ Mout) {
  __shared__ __align__(64) _Float16 Qs[64 * 64];      // [i][d]
  __shared__ __align__(64) _Float16 Ks[2][64 * 64];   // [j][d] ping-pong
  __shared__ __align__(64) _Float16 Vt[2][64 * 64];   // [d][j] ping-pong
  __shared__ __align__(64) _Float16 Ps[64 * 64];      // [i][j]

  const int bh = blockIdx.y;            // 0..15  (b*4 + h)
  const int i0 = blockIdx.x * 64;
  const int tid = threadIdx.x, wave = tid >> 5, lane = tid & 31;

  const _Float16* Qp  = Qh  + (size_t)bh * 2048 * 64;
  const _Float16* Kp  = Kh  + (size_t)bh * 2048 * 64;
  const _Float16* Vtp = Vth + (size_t)bh * 64 * 2048;

  // per-thread staging coords: 4 rows x 16B per 64x64 tile
  const int sr = tid >> 3;             // 0..15
  const int s8 = (tid & 7) << 3;       // 0..56
  unsigned ldsK0[4], ldsK1[4], ldsV0[4], ldsV1[4];
  const _Float16 *Kg[4], *Vg[4];
#pragma unroll
  for (int i = 0; i < 4; ++i) {
    int row = sr + 16 * i;
    ldsK0[i] = lds_off(&Ks[0][row * 64 + s8]);
    ldsK1[i] = lds_off(&Ks[1][row * 64 + s8]);
    ldsV0[i] = lds_off(&Vt[0][row * 64 + s8]);
    ldsV1[i] = lds_off(&Vt[1][row * 64 + s8]);
    Kg[i] = &Kp[(size_t)row * 64 + s8];          // + j0*64
    Vg[i] = &Vtp[(size_t)row * 2048 + s8];       // + j0
  }

  // prologue: Q tile + K/V tile 0 -> buffer 0
#pragma unroll
  for (int i = 0; i < 4; ++i) {
    async_copy16(lds_off(&Qs[(sr + 16 * i) * 64 + s8]),
                 &Qp[(size_t)(i0 + sr + 16 * i) * 64 + s8]);
    async_copy16(ldsK0[i], Kg[i]);
    async_copy16(ldsV0[i], Vg[i]);
  }
  wait_async<0>();
  __syncthreads();

  v16h aq0 = load_a(&Qs[(wave * 16) * 64 + 0], 64, lane);
  v16h aq1 = load_a(&Qs[(wave * 16) * 64 + 32], 64, lane);

  v8f o[4] = {};
  float mrow[8], lrow[8];
#pragma unroll
  for (int r = 0; r < 8; ++r) { mrow[r] = -1e30f; lrow[r] = 0.f; }

  const int prow = wave * 16 + ((lane >> 4) << 3);
  const int cl   = lane & 15;

  for (int jt = 0; jt < 32; jt += 2) {
    // phase A: compute buf0 (tile jt), prefetch tile jt+1 -> buf1
    flash_tile<true>(Ks[0], Vt[0], ldsK1, ldsV1, Kg, Vg,
                     (size_t)(jt + 1) << 6, aq0, aq1, o, mrow, lrow,
                     Ps, wave, lane, prow, cl);
    // phase B: compute buf1 (tile jt+1), prefetch tile jt+2 -> buf0 if any
    if (jt + 2 < 32)
      flash_tile<true>(Ks[1], Vt[1], ldsK0, ldsV0, Kg, Vg,
                       (size_t)(jt + 2) << 6, aq0, aq1, o, mrow, lrow,
                       Ps, wave, lane, prow, cl);
    else
      flash_tile<false>(Ks[1], Vt[1], ldsK0, ldsV0, Kg, Vg,
                        0, aq0, aq1, o, mrow, lrow,
                        Ps, wave, lane, prow, cl);
  }

  // store merged-head: Mout[(b*2048 + n)*256 + h*64 + dh]
  const int b = bh >> 2, h = bh & 3;
  const int rbase = i0 + wave * 16 + ((lane >> 4) << 3);
#pragma unroll
  for (int cb = 0; cb < 4; ++cb)
#pragma unroll
    for (int r = 0; r < 8; ++r) {
      float val = o[cb][r] / lrow[r];
      int n  = rbase + r;
      int dh = cb * 16 + cl;
      Mout[((size_t)(b * 2048 + n)) * 256 + h * 64 + dh] = (_Float16)val;
    }
}

// ---------------------------------------------------------------------------
// LayerNorm(512) + exact GELU, f32 in -> f16 out. One block (256 thr) per row.
// ---------------------------------------------------------------------------
__global__ __launch_bounds__(256) void ln_gelu(const float* __restrict__ hin,
                                               const float* __restrict__ g,
                                               const float* __restrict__ bb,
                                               _Float16* __restrict__ hout) {
  __shared__ float red[256];
  const int row = blockIdx.x;
  const int tid = threadIdx.x;
  float x0 = hin[(size_t)row * 512 + tid];
  float x1 = hin[(size_t)row * 512 + 256 + tid];

  red[tid] = x0 + x1;
  __syncthreads();
  for (int s = 128; s > 0; s >>= 1) {
    if (tid < s) red[tid] += red[tid + s];
    __syncthreads();
  }
  float mu = red[0] * (1.f / 512.f);
  __syncthreads();

  float d0 = x0 - mu, d1 = x1 - mu;
  red[tid] = d0 * d0 + d1 * d1;
  __syncthreads();
  for (int s = 128; s > 0; s >>= 1) {
    if (tid < s) red[tid] += red[tid + s];
    __syncthreads();
  }
  float inv = rsqrtf(red[0] * (1.f / 512.f) + 1e-5f);

  float t0 = d0 * inv * g[tid] + bb[tid];
  float t1 = d1 * inv * g[256 + tid] + bb[256 + tid];
  float g0 = 0.5f * t0 * (1.f + erff(t0 * 0.70710678118654752f));
  float g1 = 0.5f * t1 * (1.f + erff(t1 * 0.70710678118654752f));
  hout[(size_t)row * 512 + tid]       = (_Float16)g0;
  hout[(size_t)row * 512 + 256 + tid] = (_Float16)g1;
}

// ---------------------------------------------------------------------------
extern "C" void kernel_launch(void* const* d_in, const int* in_sizes, int n_in,
                              void* d_out, int out_size, void* d_ws, size_t ws_size,
                              hipStream_t stream) {
  const float* x0   = (const float*)d_in[0];
  const float* x1   = (const float*)d_in[1];
  const float* Wqk  = (const float*)d_in[2];
  const float* bqk  = (const float*)d_in[3];
  const float* Wv   = (const float*)d_in[4];
  const float* bv   = (const float*)d_in[5];
  const float* Wout = (const float*)d_in[6];
  const float* bout = (const float*)d_in[7];
  const float* Wf1  = (const float*)d_in[8];
  const float* bf1  = (const float*)d_in[9];
  const float* lng  = (const float*)d_in[10];
  const float* lnb  = (const float*)d_in[11];
  const float* Wf2  = (const float*)d_in[12];
  const float* bf2  = (const float*)d_in[13];

  const int M = 8192;  // B*N rows

  char* ws = (char*)d_ws;
  size_t off = 0;
  auto alloc = [&](size_t bytes) -> void* {
    void* p = ws + off;
    off += (bytes + 255) & ~(size_t)255;
    return p;
  };
  _Float16* cat0  = (_Float16*)alloc((size_t)M * 512 * 2);  // [x0 | m0@Wout] f16
  _Float16* cat1  = (_Float16*)alloc((size_t)M * 512 * 2);
  _Float16* qkh0  = (_Float16*)alloc((size_t)16 * 2048 * 64 * 2);  // [B*H,N,DH]
  _Float16* qkh1  = (_Float16*)alloc((size_t)16 * 2048 * 64 * 2);
  _Float16* vth0  = (_Float16*)alloc((size_t)16 * 64 * 2048 * 2); // [B*H,DH,N]
  _Float16* vth1  = (_Float16*)alloc((size_t)16 * 64 * 2048 * 2);
  _Float16* mcat0 = (_Float16*)alloc((size_t)M * 256 * 2);  // merged attn out
  _Float16* mcat1 = (_Float16*)alloc((size_t)M * 256 * 2);
  _Float16* WqkT  = (_Float16*)alloc(256 * 256 * 2);        // [N,K] f16
  _Float16* WvT   = (_Float16*)alloc(256 * 256 * 2);
  _Float16* WoT   = (_Float16*)alloc(256 * 256 * 2);
  _Float16* Wf1T  = (_Float16*)alloc(512 * 512 * 2);
  _Float16* Wf2T  = (_Float16*)alloc((size_t)256 * 512 * 2);
  float*    hbuf  = (float*)alloc((size_t)M * 512 * 4);     // FFN hidden f32
  _Float16* hg    = (_Float16*)alloc((size_t)M * 512 * 2);  // post LN+GELU f16

  const float ss = 0.35355339059327373f;  // sqrt(DH^-0.5)
  dim3 blk128(128);

  // ---- fp32 -> f16 conversions (weights pre-transposed to [N,K]) ----
  cvt_f32_f16<<<512, 256, 0, stream>>>(x0, cat0, M, 256, 512);
  cvt_f32_f16<<<512, 256, 0, stream>>>(x1, cat1, M, 256, 512);
  cvt_f32_f16_T<<<64, 256, 0, stream>>>(Wqk, WqkT, 256, 256);
  cvt_f32_f16_T<<<64, 256, 0, stream>>>(Wv, WvT, 256, 256);
  cvt_f32_f16_T<<<64, 256, 0, stream>>>(Wout, WoT, 256, 256);
  cvt_f32_f16_T<<<256, 256, 0, stream>>>(Wf1, Wf1T, 512, 512);
  cvt_f32_f16_T<<<128, 256, 0, stream>>>(Wf2, Wf2T, 512, 256);

  // ---- QK / V projections (head-split f16; qk scaled by ss; V transposed) --
  wgemm<<<dim3(4, 128), blk128, 0, stream>>>(cat0, 512, WqkT, 256, bqk, 256,
                                             2, ss, qkh0, 0, 0, nullptr);
  wgemm<<<dim3(4, 128), blk128, 0, stream>>>(cat1, 512, WqkT, 256, bqk, 256,
                                             2, ss, qkh1, 0, 0, nullptr);
  wgemm<<<dim3(4, 128), blk128, 0, stream>>>(cat0, 512, WvT, 256, bv, 256,
                                             3, 1.0f, vth0, 0, 0, nullptr);
  wgemm<<<dim3(4, 128), blk128, 0, stream>>>(cat1, 512, WvT, 256, bv, 256,
                                             3, 1.0f, vth1, 0, 0, nullptr);

  // ---- cross attention: m0 = Attn(qk0, qk1, v1); m1 = Attn(qk1, qk0, v0) ---
  flashattn<<<dim3(32, 16), blk128, 0, stream>>>(qkh0, qkh1, vth1, mcat0);
  flashattn<<<dim3(32, 16), blk128, 0, stream>>>(qkh1, qkh0, vth0, mcat1);

  // ---- output projection -> second half of concat buffers (f16) ----
  wgemm<<<dim3(4, 128), blk128, 0, stream>>>(mcat0, 256, WoT, 256, bout, 256,
                                             1, 1.0f, cat0, 512, 256, nullptr);
  wgemm<<<dim3(4, 128), blk128, 0, stream>>>(mcat1, 256, WoT, 256, bout, 256,
                                             1, 1.0f, cat1, 512, 256, nullptr);

  // ---- FFN stream 0 ----
  wgemm<<<dim3(8, 128), blk128, 0, stream>>>(cat0, 512, Wf1T, 512, bf1, 512,
                                             0, 1.0f, hbuf, 512, 0, nullptr);
  ln_gelu<<<M, 256, 0, stream>>>(hbuf, lng, lnb, hg);
  wgemm<<<dim3(4, 128), blk128, 0, stream>>>(hg, 512, Wf2T, 512, bf2, 512,
                                             0, 1.0f, (float*)d_out, 256, 0, x0);

  // ---- FFN stream 1 ----
  wgemm<<<dim3(8, 128), blk128, 0, stream>>>(cat1, 512, Wf1T, 512, bf1, 512,
                                             0, 1.0f, hbuf, 512, 0, nullptr);
  ln_gelu<<<M, 256, 0, stream>>>(hbuf, lng, lnb, hg);
  wgemm<<<dim3(4, 128), blk128, 0, stream>>>(hg, 512, Wf2T, 512, bf2, 512,
                                             0, 1.0f, (float*)d_out + (size_t)M * 256,
                                             256, 0, x1);
}
